// TopKRouter_84817014161792
// MI455X (gfx1250) — matrix operand order repaired and checked by reading, not compiled, not run on hardware
//
#include <hip/hip_runtime.h>
#include <hip/hip_bf16.h>

// ---- types matching the gfx1250 WMMA builtin signatures ----
typedef __attribute__((ext_vector_type(16))) __bf16 v16bf;
typedef __attribute__((ext_vector_type(8)))  __bf16 v8bf;
typedef __attribute__((ext_vector_type(4)))  __bf16 bf16x4;
typedef __attribute__((ext_vector_type(8)))  float  v8f;
typedef __attribute__((ext_vector_type(4)))  float  f32x4;

#define D_MODEL            2048
#define N_EXPERTS          64
#define KCHUNK             256                     // K staged in LDS per step
#define LDS_ROW            (KCHUNK + 8)            // 264 bf16 = 528 B (4-bank row skew)
#define WAVES_PER_BLOCK    8
#define TOKENS_PER_WAVE    16
#define TOKENS_PER_BLOCK   (WAVES_PER_BLOCK * TOKENS_PER_WAVE)  // 128

__global__ __launch_bounds__(256)
void topk_router_wmma(const float* __restrict__ x,
                      const float* __restrict__ W,
                      float* __restrict__ gates_out,
                      int*   __restrict__ idx_out)
{
    // Reused: bf16 W chunk (64*528 = 33792 B) then per-wave logits (8*16*64*4 = 32768 B)
    __shared__ __align__(16) unsigned char smemRaw[N_EXPERTS * LDS_ROW * 2];
    __bf16* Wlds = (__bf16*)smemRaw;
    float*  Llds = (float*)smemRaw;

    const int tid  = threadIdx.x;
    const int lane = tid & 31;
    const int wave = tid >> 5;
    const int hf   = lane >> 4;      // half-wave: 0 -> lanes 0-15, 1 -> lanes 16-31
    const int l15  = lane & 15;

    const int tokWave = blockIdx.x * TOKENS_PER_BLOCK + wave * TOKENS_PER_WAVE;

    // A-fragment source row for this lane (token m = tokWave + l15)
    const float* xrow = x + (size_t)(tokWave + l15) * D_MODEL;
    const int akb = hf * 16;         // A K-sub-base inside each 32-wide step

    v8f acc0 = {}, acc1 = {}, acc2 = {}, acc3 = {};

    // Cooperative W staging geometry: 64 rows x KCHUNK f32, 4 threads per row
    const int wRow = tid >> 2;       // 0..63 (expert)
    const int wSub = tid & 3;

    for (int kc = 0; kc < D_MODEL; kc += KCHUNK) {
        // ---- stage W[:, kc : kc+KCHUNK] into LDS as bf16 ----
        {
            const float* wsrc = W + (size_t)wRow * D_MODEL + kc + wSub * 4;
            __bf16*      wdst = Wlds + wRow * LDS_ROW + wSub * 4;
            #pragma unroll
            for (int i = 0; i < KCHUNK / 16; ++i) {
                f32x4 v = *(const f32x4*)(wsrc + i * 16);
                bf16x4 o;
                o[0] = (__bf16)v[0]; o[1] = (__bf16)v[1];
                o[2] = (__bf16)v[2]; o[3] = (__bf16)v[3];
                *(bf16x4*)(wdst + i * 16) = o;
            }
        }
        __syncthreads();

        // stream-prefetch next K-chunk of this lane's token row
        if (kc + KCHUNK < D_MODEL)
            __builtin_prefetch(xrow + kc + KCHUNK, 0, 3);

        // ---- consume: KCHUNK/32 WMMA k-steps ----
        #pragma unroll
        for (int ks = 0; ks < KCHUNK; ks += 32) {
            // A fragment: lanes 0-15 K=0..15, lanes 16-31 K=16..31
            const float* ap = xrow + kc + ks + akb;
            f32x4 a0 = *(const f32x4*)(ap + 0);
            f32x4 a1 = *(const f32x4*)(ap + 4);
            f32x4 a2 = *(const f32x4*)(ap + 8);
            f32x4 a3 = *(const f32x4*)(ap + 12);

            // B fragments: lane holds expert col n = 16*t + l15,
            // K = ks + 16*hf + [0..15]. Load ALL FOUR tiles into distinct
            // registers first so the ds_load_b128s issue as one clause and
            // their latency overlaps with the WMMA sequence.
            const __bf16* bbase = Wlds + l15 * LDS_ROW + ks + 16 * hf;
            v8bf b0l = *(const v8bf*)(bbase + 0 * 16 * LDS_ROW + 0);
            v8bf b0h = *(const v8bf*)(bbase + 0 * 16 * LDS_ROW + 8);
            v8bf b1l = *(const v8bf*)(bbase + 1 * 16 * LDS_ROW + 0);
            v8bf b1h = *(const v8bf*)(bbase + 1 * 16 * LDS_ROW + 8);
            v8bf b2l = *(const v8bf*)(bbase + 2 * 16 * LDS_ROW + 0);
            v8bf b2h = *(const v8bf*)(bbase + 2 * 16 * LDS_ROW + 8);
            v8bf b3l = *(const v8bf*)(bbase + 3 * 16 * LDS_ROW + 0);
            v8bf b3h = *(const v8bf*)(bbase + 3 * 16 * LDS_ROW + 8);

            v16bf a;
            #pragma unroll
            for (int i = 0; i < 4; ++i) {
                a[i]      = (__bf16)a0[i];
                a[4 + i]  = (__bf16)a1[i];
                a[8 + i]  = (__bf16)a2[i];
                a[12 + i] = (__bf16)a3[i];
            }

            v16bf b0, b1, b2, b3;
            #pragma unroll
            for (int i = 0; i < 8; ++i) {
                b0[i] = b0l[i]; b0[8 + i] = b0h[i];
                b1[i] = b1l[i]; b1[8 + i] = b1h[i];
                b2[i] = b2l[i]; b2[8 + i] = b2h[i];
                b3[i] = b3l[i]; b3[8 + i] = b3h[i];
            }

            acc0 = __builtin_amdgcn_wmma_f32_16x16x32_bf16(
                       false, a, false, b0, (short)0, acc0, false, false);
            acc1 = __builtin_amdgcn_wmma_f32_16x16x32_bf16(
                       false, a, false, b1, (short)0, acc1, false, false);
            acc2 = __builtin_amdgcn_wmma_f32_16x16x32_bf16(
                       false, a, false, b2, (short)0, acc2, false, false);
            acc3 = __builtin_amdgcn_wmma_f32_16x16x32_bf16(
                       false, a, false, b3, (short)0, acc3, false, false);
        }
        __syncthreads();   // protect W chunk until all waves consumed it
    }

    // ---- epilogue: scatter logits to LDS so each token is contiguous ----
    float* myL = Llds + wave * (TOKENS_PER_WAVE * N_EXPERTS);
    #pragma unroll
    for (int r = 0; r < 8; ++r) {
        const int m = r + 8 * hf;                 // token within wave tile
        myL[m * N_EXPERTS +  0 + l15] = acc0[r];
        myL[m * N_EXPERTS + 16 + l15] = acc1[r];
        myL[m * N_EXPERTS + 32 + l15] = acc2[r];
        myL[m * N_EXPERTS + 48 + l15] = acc3[r];
    }
    __syncthreads();

    // lanes 0-15: one token each -> top-2 + renormalized softmax (denominator cancels)
    if (lane < 16) {
        const float* lg = myL + lane * N_EXPERTS;
        float best1 = -__builtin_inff(), best2 = -__builtin_inff();
        int i1 = 0, i2 = 0;
        #pragma unroll
        for (int e = 0; e < N_EXPERTS; e += 4) {
            f32x4 v = *(const f32x4*)(lg + e);
            #pragma unroll
            for (int j = 0; j < 4; ++j) {
                const float val = v[j];
                if (val > best1) { best2 = best1; i2 = i1; best1 = val; i1 = e + j; }
                else if (val > best2) { best2 = val; i2 = e + j; }
            }
        }
        const float g1 = 1.0f / (1.0f + __expf(best2 - best1));
        const int gt = tokWave + lane;
        gates_out[gt * 2 + 0] = g1;
        gates_out[gt * 2 + 1] = 1.0f - g1;
        idx_out[gt * 2 + 0] = i1;
        idx_out[gt * 2 + 1] = i2;
    }
}

extern "C" void kernel_launch(void* const* d_in, const int* in_sizes, int n_in,
                              void* d_out, int out_size, void* d_ws, size_t ws_size,
                              hipStream_t stream) {
    (void)n_in; (void)out_size; (void)d_ws; (void)ws_size;
    const float* x = (const float*)d_in[0];    // [4,4096,2048] f32
    const float* W = (const float*)d_in[1];    // [64,2048] f32
    const int tokens = in_sizes[0] / D_MODEL;  // 16384

    float* gates = (float*)d_out;                          // [tokens,2] f32
    int*   idx   = (int*)d_out + (size_t)tokens * 2;       // [tokens,2] i32, after gates

    dim3 grid(tokens / TOKENS_PER_BLOCK);                  // 128 blocks
    topk_router_wmma<<<grid, 256, 0, stream>>>(x, W, gates, idx);
}
